// ACE_25254407700951
// MI455X (gfx1250) — matrix operand order repaired and checked by reading, not compiled, not run on hardware
//
#include <hip/hip_runtime.h>
#include <hip/hip_bf16.h>

#define Bn   2
#define Cn   512
#define Hn   128
#define Wn   128
#define NCLS 19
#define SLn  512
#define NHID 128
#define HP   130          // padded spatial (zero halo)
#define KDIM 1152         // 9 taps * 128 channels
#define MROW 1024         // interleaved gamma/beta out rows
#define PATCH_W 34        // 32-pixel tile + halo
#define PATCH_ROW (PATCH_W * NHID)        // ushorts per patch row
#define PATCH_CHUNKS (3 * PATCH_ROW * 2 / 16)  // 16B chunks in patch = 1632

typedef __attribute__((ext_vector_type(16))) __bf16 v16bf;
typedef __attribute__((ext_vector_type(8)))  float  v8f;

union Frag { v16bf v; int4 q[2]; };

__device__ __forceinline__ unsigned short f2bf(float f) {
    unsigned u = __float_as_uint(f);
    unsigned r = u + 0x7FFFu + ((u >> 16) & 1u);
    return (unsigned short)(r >> 16);
}
__device__ __forceinline__ float sigmoidf(float v) {
    return 1.0f / (1.0f + __expf(-v));
}

// CDNA5 async global->LDS copy (16B per lane), tracked by ASYNCcnt.
__device__ __forceinline__ void async_copy16(unsigned lds_off, const void* gaddr) {
    asm volatile("global_load_async_to_lds_b128 %0, %1, off"
                 :: "v"(lds_off), "v"((unsigned long long)gaddr)
                 : "memory");
}
__device__ __forceinline__ void wait_async0() {
    asm volatile("s_wait_asynccnt 0x0" ::: "memory");
}

// ---------------- labels: padded argmax(one-hot) map, border = sentinel 19 --
__global__ void k_labels(const float* __restrict__ segmap, int* __restrict__ lab) {
    int idx = blockIdx.x * 256 + threadIdx.x;
    if (idx >= Bn * HP * HP) return;
    int b = idx / (HP * HP);
    int rem = idx % (HP * HP);
    int hp = rem / HP, wp = rem % HP;
    int j = NCLS;  // sentinel: zero contribution
    if (hp >= 1 && hp <= Hn && wp >= 1 && wp <= Wn) {
        int h = hp - 1, w = wp - 1;
        for (int t = 0; t < NCLS; t++) {
            if (segmap[(((size_t)b * NCLS + t) * Hn + h) * Wn + w] > 0.5f) { j = t; break; }
        }
    }
    lab[idx] = j;
}

// ---------------- per-class FC + relu: mu_codes[b][j][d] ---------------------
__global__ void k_mu(const float* __restrict__ style, const float* __restrict__ Wmu,
                     const float* __restrict__ bmu, float* __restrict__ mu) {
    int bj = blockIdx.x;            // 0..37
    int b = bj / NCLS, j = bj % NCLS;
    __shared__ float s[SLn];
    for (int c = threadIdx.x; c < SLn; c += 256)
        s[c] = style[((size_t)b * NCLS + j) * SLn + c];
    __syncthreads();
    const float* Wj = Wmu + (size_t)j * SLn * SLn;
    for (int d = threadIdx.x; d < SLn; d += 256) {
        float acc = bmu[j * SLn + d];
        for (int c = 0; c < SLn; c++) acc = fmaf(s[c], Wj[(size_t)c * SLn + d], acc);
        mu[((size_t)b * NCLS + j) * SLn + d] = fmaxf(acc, 0.0f);
    }
}

// ---------------- instance-norm stats per (b,c): mean, rstd -----------------
__global__ void k_stats(const float* __restrict__ x, const float* __restrict__ noise,
                        const float* __restrict__ nvar, float* __restrict__ stats) {
    int bc = blockIdx.x;                  // 0..1023
    int b = bc >> 9;
    int c = bc & (Cn - 1);
    int tid = threadIdx.x;
    float nv = nvar[c];
    const float* xp = x + (size_t)bc * (Hn * Wn);
    const float* np = noise + (size_t)b * (Wn * Hn);
    float s = 0.f, s2 = 0.f;
    for (int i = tid; i < Hn * Wn; i += 256) {
        int h = i >> 7, w = i & 127;
        float v = xp[i] + np[w * Hn + h] * nv;   // noise[b,w,h,0]
        s += v; s2 += v * v;
    }
    __shared__ float rs[256], rq[256];
    rs[tid] = s; rq[tid] = s2;
    __syncthreads();
    for (int o = 128; o > 0; o >>= 1) {
        if (tid < o) { rs[tid] += rs[tid + o]; rq[tid] += rq[tid + o]; }
        __syncthreads();
    }
    if (tid == 0) {
        float m = rs[0] * (1.0f / 16384.0f);
        float var = rq[0] * (1.0f / 16384.0f) - m * m;
        stats[bc * 2 + 0] = m;
        stats[bc * 2 + 1] = rsqrtf(var + 1e-5f);
    }
}

// ---------------- Gtab[b][j][tap][{g,b}·512] = sum_d mu * w[c,d,tap] --------
__global__ void k_gtab(const float* __restrict__ mu, const float* __restrict__ wg,
                       const float* __restrict__ wb, float* __restrict__ gtab) {
    int blk = blockIdx.x;                 // (b*19+j)*9+tap, 342 blocks, 512 thr
    int tap = blk % 9;
    int bj = blk / 9;
    int b = bj / NCLS, j = bj % NCLS;
    __shared__ float m[SLn];
    int tid = threadIdx.x;
    m[tid] = mu[((size_t)b * NCLS + j) * SLn + tid];
    __syncthreads();
    int c = tid;
    float g = 0.f, bt = 0.f;
    for (int d = 0; d < SLn; d++) {
        float mv = m[d];
        g  = fmaf(mv, wg[((size_t)c * SLn + d) * 9 + tap], g);
        bt = fmaf(mv, wb[((size_t)c * SLn + d) * 9 + tap], bt);
    }
    float* dst = gtab + ((size_t)((b * (NCLS + 1) + j) * 9 + tap)) * (2 * Cn);
    dst[c] = g;
    dst[Cn + c] = bt;
}

// ---------------- padded actv (19->128 conv via label gather) bf16 NHWC -----
__global__ void k_actv(const int* __restrict__ lab, const float* __restrict__ sw,
                       const float* __restrict__ sb, unsigned short* __restrict__ actvp) {
    int p = blockIdx.x * 2 + (threadIdx.x >> 7);     // pixel id over B*H*W
    int n = threadIdx.x & (NHID - 1);
    int b = p >> 14;
    int rem = p & 16383;
    int h = rem >> 7, w = rem & 127;
    float acc = sb[n];
#pragma unroll
    for (int tap = 0; tap < 9; tap++) {
        int j = lab[(b * HP + h + tap / 3) * HP + (w + tap % 3)];
        if (j < NCLS) acc += sw[((size_t)n * NCLS + j) * 9 + tap];
    }
    acc = fmaxf(acc, 0.0f);
    actvp[((size_t)(b * HP + h + 1) * HP + (w + 1)) * NHID + n] = f2bf(acc);
}

// ---------------- pack interleaved (2c|2c+1 = gamma|beta) bf16 weights ------
__global__ void k_pack(const float* __restrict__ wg, const float* __restrict__ wb,
                       unsigned short* __restrict__ wcat) {
    int idx = blockIdx.x * 256 + threadIdx.x;
    if (idx >= MROW * KDIM) return;
    int row = idx / KDIM;
    int col = idx % KDIM;
    int tap = col / NHID, n = col % NHID;
    int c = row >> 1;
    const float* src = (row & 1) ? wb : wg;
    wcat[idx] = f2bf(src[((size_t)c * NHID + n) * 9 + tap]);
}

// ---------------- SPADE conv via WMMA implicit GEMM + fused epilogue --------
// Block: 32-pixel tile x 512 interleaved rows. im2col patch async-staged in LDS.
// Writes P2 = norm*(1-ga)*gsp + (1-ba)*bsp to out.
__global__ void __launch_bounds__(256)
k_spade_wmma(const unsigned short* __restrict__ actvp, const unsigned short* __restrict__ wcat,
             const float* __restrict__ x, const float* __restrict__ noise,
             const float* __restrict__ nvar, const float* __restrict__ stats,
             const float* __restrict__ gbias, const float* __restrict__ bbias,
             const float* __restrict__ bg, const float* __restrict__ bb,
             float* __restrict__ out) {
    __shared__ unsigned short sB[3 * PATCH_ROW];     // 3 x 34 x 128 bf16 = 26112 B
    int tile = blockIdx.x;                    // B*H*(W/32) = 1024
    int wtile = tile & 3;
    int h = (tile >> 2) & 127;
    int b = tile >> 9;
    int w0 = wtile * 32;
    int tid = threadIdx.x;
    int wave = tid >> 5;
    int lane = tid & 31;
    int lhalf = lane >> 4;
    int l16 = lane & 15;
    int mbase = blockIdx.y * 512 + wave * 64;  // 64 interleaved rows per wave

    // Async-stage the 3-row x 34-pixel x 128-ch actv patch (each row contiguous).
    unsigned sbase = (unsigned)(size_t)(&sB[0]);
    for (int s = tid; s < PATCH_CHUNKS; s += 256) {
        int rr = s / (PATCH_CHUNKS / 3);
        int ck = s % (PATCH_CHUNKS / 3);          // 16B chunk within row
        const char* g = (const char*)(actvp + ((size_t)(b * HP + h + rr) * HP + w0) * NHID)
                        + ck * 16;
        async_copy16(sbase + (unsigned)(rr * PATCH_ROW * 2 + ck * 16), g);
    }
    wait_async0();
    __syncthreads();

    v8f acc[4][2] = {};

    for (int kk = 0; kk < KDIM; kk += 32) {
        int tap = kk >> 7;                     // 128 | kk-block, so tap is uniform
        int kin = kk & 127;
        int ody = tap / 3, odx = tap % 3;
        // B fragments from LDS: lane holds column N=l16 (pixel),
        // K elems kk+lhalf*16+0..15
        Frag bf0, bf1;
        {
            const int4* p0 = (const int4*)&sB[ody * PATCH_ROW + (l16 + odx) * NHID + kin + lhalf * 16];
            bf0.q[0] = p0[0]; bf0.q[1] = p0[1];
            const int4* p1 = (const int4*)&sB[ody * PATCH_ROW + (16 + l16 + odx) * NHID + kin + lhalf * 16];
            bf1.q[0] = p1[0]; bf1.q[1] = p1[1];
        }
#pragma unroll
        for (int a = 0; a < 4; a++) {
            // A fragment: lane holds row M=mbase+a*16+l16;
            // elems e0..15 = K kk+lhalf*16+0..15
            const unsigned short* ap = wcat + (size_t)(mbase + a * 16 + l16) * KDIM + kk + lhalf * 16;
            Frag af;
            af.q[0] = *(const int4*)(ap);
            af.q[1] = *(const int4*)(ap + 8);
            acc[a][0] = __builtin_amdgcn_wmma_f32_16x16x32_bf16(
                false, af.v, false, bf0.v, (short)0, acc[a][0], false, false);
            acc[a][1] = __builtin_amdgcn_wmma_f32_16x16x32_bf16(
                false, af.v, false, bf1.v, (short)0, acc[a][1], false, false);
        }
    }

    float ga = sigmoidf(bg[0]);
    float ba = sigmoidf(bb[0]);
    float omg = 1.0f - ga, omb = 1.0f - ba;
#pragma unroll
    for (int nt = 0; nt < 2; nt++) {
        int w = w0 + nt * 16 + l16;
        float nz = noise[((size_t)b * Wn + w) * Hn + h];   // noise[b,w,h,0]
#pragma unroll
        for (int a = 0; a < 4; a++) {
#pragma unroll
            for (int i = 0; i < 4; i++) {
                int r = 2 * i;
                int row = mbase + a * 16 + lhalf * 8 + r;  // even row -> gamma channel
                int c = row >> 1;
                float g  = acc[a][nt][r]     + gbias[c];
                float bt = acc[a][nt][r + 1] + bbias[c];
                size_t idx = (((size_t)b * Cn + c) * Hn + h) * Wn + w;
                int sc = (b * Cn + c) * 2;
                float norm = (x[idx] + nz * nvar[c] - stats[sc]) * stats[sc + 1];
                out[idx] = norm * omg * g + omb * bt;
            }
        }
    }
}

// ---------------- avg branch: Gtab gather + blend, out += P1 ----------------
__global__ void __launch_bounds__(256)
k_avg_out(const float* __restrict__ x, const float* __restrict__ noise,
          const float* __restrict__ nvar, const float* __restrict__ stats,
          const float* __restrict__ gtab, const int* __restrict__ lab,
          const float* __restrict__ cgb, const float* __restrict__ cbb,
          const float* __restrict__ bg, const float* __restrict__ bb,
          float* __restrict__ out) {
    __shared__ float lg[8][520];
    __shared__ float lb[8][520];
    __shared__ int labs[3][10];
    int tile = blockIdx.x;                    // B*H*(W/8) = 4096
    int wtile = tile & 15;
    int h = (tile >> 4) & 127;
    int b = tile >> 11;
    int w0 = wtile * 8;
    int tid = threadIdx.x;
    if (tid < 30) {
        int rr = tid / 10, cc = tid % 10;
        labs[rr][cc] = lab[(b * HP + h + rr) * HP + (w0 + cc)];
    }
    __syncthreads();
    for (int px = 0; px < 8; px++) {
        for (int c = tid; c < Cn; c += 256) {
            float g = cgb[c], bt = cbb[c];
#pragma unroll
            for (int tap = 0; tap < 9; tap++) {
                int j = labs[tap / 3][px + tap % 3];
                const float* base =
                    gtab + ((size_t)((b * (NCLS + 1) + j) * 9 + tap)) * (2 * Cn);
                g  += base[c];
                bt += base[Cn + c];
            }
            lg[px][c] = g;
            lb[px][c] = bt;
        }
    }
    __syncthreads();
    float ga = sigmoidf(bg[0]);
    float ba = sigmoidf(bb[0]);
    int wl = tid & 7;
    int crow = tid >> 3;                      // 32 channels per pass
    int w = w0 + wl;
    float nz = noise[((size_t)b * Wn + w) * Hn + h];
    for (int pass = 0; pass < 16; pass++) {
        int c = pass * 32 + crow;
        int sc = (b * Cn + c) * 2;
        size_t idx = (((size_t)b * Cn + c) * Hn + h) * Wn + w;
        float norm = (x[idx] + nz * nvar[c] - stats[sc]) * stats[sc + 1];
        out[idx] += norm * (1.0f + ga * lg[wl][c]) + ba * lb[wl][c];
    }
}

// ---------------------------------------------------------------------------
static inline size_t alignup(size_t v) { return (v + 255) & ~(size_t)255; }

extern "C" void kernel_launch(void* const* d_in, const int* in_sizes, int n_in,
                              void* d_out, int out_size, void* d_ws, size_t ws_size,
                              hipStream_t stream) {
    const float* x        = (const float*)d_in[0];
    const float* segmap   = (const float*)d_in[1];
    const float* style    = (const float*)d_in[2];
    const float* noise    = (const float*)d_in[3];
    const float* nvar     = (const float*)d_in[4];
    const float* bg       = (const float*)d_in[5];
    const float* bb       = (const float*)d_in[6];
    const float* Wmu      = (const float*)d_in[7];
    const float* bmu      = (const float*)d_in[8];
    const float* cgw      = (const float*)d_in[9];
    const float* cgb      = (const float*)d_in[10];
    const float* cbw      = (const float*)d_in[11];
    const float* cbb      = (const float*)d_in[12];
    const float* spsw     = (const float*)d_in[13];
    const float* spsb     = (const float*)d_in[14];
    const float* spgw     = (const float*)d_in[15];
    const float* spgb     = (const float*)d_in[16];
    const float* spbw     = (const float*)d_in[17];
    const float* spbb     = (const float*)d_in[18];
    float* out = (float*)d_out;

    char* ws = (char*)d_ws;
    float* mu = (float*)ws;              ws += alignup((size_t)Bn * NCLS * SLn * 4);
    int* lab = (int*)ws;                 ws += alignup((size_t)Bn * HP * HP * 4);
    float* stats = (float*)ws;           ws += alignup((size_t)Bn * Cn * 2 * 4);
    size_t gtab_bytes = (size_t)Bn * (NCLS + 1) * 9 * 2 * Cn * 4;
    float* gtab = (float*)ws;            ws += alignup(gtab_bytes);
    size_t actv_bytes = (size_t)Bn * HP * HP * NHID * 2;
    unsigned short* actvp = (unsigned short*)ws; ws += alignup(actv_bytes);
    unsigned short* wcat = (unsigned short*)ws;  ws += alignup((size_t)MROW * KDIM * 2);

    hipMemsetAsync(gtab, 0, gtab_bytes, stream);     // class-19 (sentinel) slab = 0
    hipMemsetAsync(actvp, 0, actv_bytes, stream);    // zero halo for SAME padding

    k_labels<<<(Bn * HP * HP + 255) / 256, 256, 0, stream>>>(segmap, lab);
    k_mu<<<Bn * NCLS, 256, 0, stream>>>(style, Wmu, bmu, mu);
    k_stats<<<Bn * Cn, 256, 0, stream>>>(x, noise, nvar, stats);
    k_gtab<<<Bn * NCLS * 9, 512, 0, stream>>>(mu, cgw, cbw, gtab);
    k_actv<<<Bn * Hn * Wn / 2, 256, 0, stream>>>(lab, spsw, spsb, actvp);
    k_pack<<<(MROW * KDIM + 255) / 256, 256, 0, stream>>>(spgw, spbw, wcat);
    k_spade_wmma<<<dim3(Bn * Hn * (Wn / 32), 2), 256, 0, stream>>>(
        actvp, wcat, x, noise, nvar, stats, spgb, spbb, bg, bb, out);
    k_avg_out<<<Bn * Hn * (Wn / 8), 256, 0, stream>>>(
        x, noise, nvar, stats, gtab, lab, cgb, cbb, bg, bb, out);
    (void)in_sizes; (void)n_in; (void)out_size; (void)ws_size;
}
